// Refine_Network_27238682591352
// MI455X (gfx1250) — matrix operand order repaired and checked by reading, not compiled, not run on hardware
//
#include <hip/hip_runtime.h>
#include <hip/hip_bf16.h>
#include <math.h>

// ---------------- problem constants (match reference) ----------------
#define Lres     512
#define NNODE    512          // B*L
#define DMSA     256
#define DPAIR    128
#define DSTATE   32
#define TOPK     64
#define NEDGE    (NNODE * TOPK)   // 32768
#define DNODEIN  309              // 256 + 21 + 32
#define DINV     100              // 32+32+32+1+3  (padded to 128 for WMMA)
#define EPSLN    1e-5f

typedef __attribute__((ext_vector_type(16))) _Float16 v16h;
typedef __attribute__((ext_vector_type(4)))  _Float16 v4h;
typedef __attribute__((ext_vector_type(8)))  float    v8f;

union frag16 { uint4 q[2]; v16h v; };

// ---------------- wave32 helpers ----------------
__device__ inline float wredf(float x) {          // full-wave allreduce
#pragma unroll
    for (int m = 16; m > 0; m >>= 1) x += __shfl_xor(x, m, 32);
    return x;
}
__device__ inline float hredf(float x) {          // allreduce within 16-lane half
#pragma unroll
    for (int m = 8; m > 0; m >>= 1) x += __shfl_xor(x, m, 32);
    return x;
}
__device__ inline float gelu_tanh(float x) {      // jax.nn.gelu approximate=True
    float x3 = x * x * x;
    return 0.5f * x * (1.f + tanhf(0.7978845608028654f * (x + 0.044715f * x3)));
}

// ---- WMMA fragment loaders (ISA 7.12.2 layouts, wave32), b128 LDS loads ----
// A tile is row-major [16][128] f16.  Per lane (row m = lane&15, hi = lane>>4):
//   elems 0-7  <- K = kc*32 + hi*8 + (0..7)        (16 contiguous bytes)
//   elems 8-15 <- K = kc*32 + 16 + hi*8 + (0..7)   (16 contiguous bytes)
__device__ inline v16h load_a_frag(const _Float16* a, int lane, int kc) {
    int m  = lane & 15;
    int hi = (lane >> 4) & 1;
    const _Float16* p = a + m * 128 + kc * 32 + hi * 8;
    frag16 u;
    u.q[0] = *reinterpret_cast<const uint4*>(p);
    u.q[1] = *reinterpret_cast<const uint4*>(p + 16);
    return u.v;
}

// B weights stored TRANSPOSED in LDS: wT[n][K], n=0..31, K=0..127.
// Per lane (col n = (lane&15)+n0, hi): elems 0-7 <- K = kc*32 + hi*8 + (0..7),
// elems 8-15 <- K = kc*32 + 16 + hi*8 + (0..7) -> two 16-byte runs, 16B aligned.
__device__ inline v16h load_bT_frag(const _Float16* wT, int lane, int kc, int n0) {
    int n  = (lane & 15) + n0;
    int hi = (lane >> 4) & 1;
    const _Float16* p = wT + n * 128 + kc * 32 + hi * 8;
    frag16 u;
    u.q[0] = *reinterpret_cast<const uint4*>(p);
    u.q[1] = *reinterpret_cast<const uint4*>(p + 16);
    return u.v;
}

// ---------------- kernel 1: CA and local frame vectors ----------------
__global__ void prep_kernel(const float* __restrict__ xyz,
                            float* __restrict__ CA, float* __restrict__ vv) {
    int n = blockIdx.x * blockDim.x + threadIdx.x;
    if (n >= NNODE) return;
    const float* p = xyz + n * 9;             // (3 atoms x 3)
    float cax = p[3], cay = p[4], caz = p[5]; // atom index 1 = CA
    CA[n * 3 + 0] = cax; CA[n * 3 + 1] = cay; CA[n * 3 + 2] = caz;
#pragma unroll
    for (int i = 0; i < 3; ++i) {
        vv[n * 9 + i * 3 + 0] = p[i * 3 + 0] - cax;
        vv[n * 9 + i * 3 + 1] = p[i * 3 + 1] - cay;
        vv[n * 9 + i * 3 + 2] = p[i * 3 + 2] - caz;
    }
}

// ---------------- kernel 2: node features l0 ----------------
__global__ __launch_bounds__(256)
void node_kernel(const float* __restrict__ msa, const float* __restrict__ seq1hot,
                 const float* __restrict__ state,
                 const float* __restrict__ g_msa, const float* __restrict__ b_msa,
                 const float* __restrict__ g_state, const float* __restrict__ b_state,
                 const float* __restrict__ g_node, const float* __restrict__ b_node,
                 const float* __restrict__ Wx, const float* __restrict__ bx,
                 float* __restrict__ l0) {
    __shared__ float sIn[DNODEIN];
    __shared__ float sRed[256];
    int n = blockIdx.x, t = threadIdx.x;

    float x = msa[n * DMSA + t];
    sRed[t] = x; __syncthreads();
    for (int o = 128; o > 0; o >>= 1) { if (t < o) sRed[t] += sRed[t + o]; __syncthreads(); }
    float mu = sRed[0] * (1.f / DMSA); __syncthreads();
    float d = x - mu;
    sRed[t] = d * d; __syncthreads();
    for (int o = 128; o > 0; o >>= 1) { if (t < o) sRed[t] += sRed[t + o]; __syncthreads(); }
    float var = sRed[0] * (1.f / DMSA); __syncthreads();
    sIn[t] = d * rsqrtf(var + EPSLN) * g_msa[t] + b_msa[t];

    if (t < 21) sIn[DMSA + t] = seq1hot[n * 21 + t];
    if (t < 32) {                        // LN(state) over 32 (single wave)
        float sv = state[n * DSTATE + t];
        float mu2 = wredf(sv) * (1.f / 32.f);
        float dd = sv - mu2;
        float v2 = wredf(dd * dd) * (1.f / 32.f);
        sIn[DMSA + 21 + t] = dd * rsqrtf(v2 + EPSLN) * g_state[t] + b_state[t];
    }
    __syncthreads();

    if (t < 32) {                        // x = node_in @ Wx + bx ; LN over 32
        float acc = bx[t];
        for (int k = 0; k < DNODEIN; ++k) acc += sIn[k] * Wx[k * 32 + t];
        float mu3 = wredf(acc) * (1.f / 32.f);
        float dd = acc - mu3;
        float v3 = wredf(dd * dd) * (1.f / 32.f);
        l0[n * 32 + t] = dd * rsqrtf(v3 + EPSLN) * g_node[t] + b_node[t];
    }
}

// ---------------- kernel 3: distances + top-k ----------------
__global__ __launch_bounds__(256)
void topk_kernel(const float* __restrict__ CA, int* __restrict__ nbr) {
    __shared__ float dist[Lres];
    __shared__ float sVal[256];
    __shared__ int   sIdx[256];
    int i = blockIdx.x, t = threadIdx.x;
    float cx = CA[i * 3], cy = CA[i * 3 + 1], cz = CA[i * 3 + 2];
    for (int j = t; j < Lres; j += 256) {
        float dx = CA[j * 3] - cx, dy = CA[j * 3 + 1] - cy, dz = CA[j * 3 + 2] - cz;
        float dd = sqrtf(dx * dx + dy * dy + dz * dz + 1e-8f);
        if (j == i) dd += 999.9f;
        dist[j] = dd;
    }
    __syncthreads();
    for (int k = 0; k < TOPK; ++k) {
        float bv = 3.4e38f; int bi = 0x7fffffff;
        for (int j = t; j < Lres; j += 256) {
            float dv = dist[j];
            if (dv < bv || (dv == bv && j < bi)) { bv = dv; bi = j; }
        }
        sVal[t] = bv; sIdx[t] = bi; __syncthreads();
        for (int o = 128; o > 0; o >>= 1) {
            if (t < o) {
                float ov = sVal[t + o]; int oi = sIdx[t + o];
                if (ov < sVal[t] || (ov == sVal[t] && oi < sIdx[t])) { sVal[t] = ov; sIdx[t] = oi; }
            }
            __syncthreads();
        }
        if (t == 0) { nbr[i * TOPK + k] = sIdx[0]; dist[sIdx[0]] = 3.0e38f; }
        __syncthreads();
    }
}

// ---------------- kernel 4: zero scratch ----------------
__global__ void zero_kernel(float* __restrict__ p, int n) {
    int i = blockIdx.x * blockDim.x + threadIdx.x;
    if (i < n) p[i] = 0.f;
}

// ---------------- kernel 5: fused edge stage (2x WMMA) ----------------
// 4 waves/block, 16 edges/wave, 512 blocks -> 32768 edges
#define WPB 4
__global__ __launch_bounds__(128)
void edge_kernel(const float* __restrict__ pair,
                 const float* __restrict__ g_pair, const float* __restrict__ b_pair,
                 const float* __restrict__ We1, const float* __restrict__ be1,
                 const float* __restrict__ g_e1, const float* __restrict__ b_e1,
                 const float* __restrict__ W1,  const float* __restrict__ b1w,
                 const float* __restrict__ Wv,  const float* __restrict__ Wvv,
                 const float* __restrict__ l0,  const float* __restrict__ CA,
                 const float* __restrict__ vv,  const int* __restrict__ nbr,
                 float* __restrict__ agg, float* __restrict__ v_agg) {
    __shared__ __align__(16) _Float16 sWe1T[32 * 128];  // transposed [n][K]
    __shared__ __align__(16) _Float16 sW1T [32 * 128];  // transposed [n][K], K>=100 zero
    __shared__ __align__(16) _Float16 sA   [WPB][16 * 128];
    __shared__ float    sT  [WPB][16 * 32];
    __shared__ float    sMF [WPB][16][8];   // d_e(3), dn, sv(3), pad
    __shared__ int      sMI [WPB][16][2];   // src, dst

    int t = threadIdx.x, lane = t & 31, wid = t >> 5;
    int col = lane & 15;
    int hi  = lane >> 4;

    // transpose weights into LDS as f16 (W1 padded 100 -> 128 rows of K)
    for (int idx = t; idx < 32 * 128; idx += 128) {
        int n = idx >> 7, K = idx & 127;
        sWe1T[idx] = (_Float16)We1[K * 32 + n];
        sW1T[idx]  = (K < DINV) ? (_Float16)W1[K * 32 + n] : (_Float16)0.f;
    }
    __syncthreads();

    int ebase = (blockIdx.x * WPB + wid) * 16;

    // per-edge metadata
    if (lane < 16) {
        int e   = ebase + lane;
        int src = e >> 6;
        int kk  = e & 63;
        int dst = nbr[src * TOPK + kk];
        float dx = CA[dst * 3 + 0] - CA[src * 3 + 0];
        float dy = CA[dst * 3 + 1] - CA[src * 3 + 1];
        float dz = CA[dst * 3 + 2] - CA[src * 3 + 2];
        float dn = sqrtf(dx * dx + dy * dy + dz * dz);
        float inv = 1.f / (dn + 1e-8f);
        float ux = dx * inv, uy = dy * inv, uz = dz * inv;
        const float* vs = vv + src * 9;
        sMI[wid][lane][0] = src; sMI[wid][lane][1] = dst;
        sMF[wid][lane][0] = dx;  sMF[wid][lane][1] = dy; sMF[wid][lane][2] = dz;
        sMF[wid][lane][3] = dn;
        sMF[wid][lane][4] = vs[0] * ux + vs[1] * uy + vs[2] * uz;
        sMF[wid][lane][5] = vs[3] * ux + vs[4] * uy + vs[5] * uz;
        sMF[wid][lane][6] = vs[6] * ux + vs[7] * uy + vs[8] * uz;
    }
    __syncthreads();

    // per-lane LN params for the pair LayerNorm (vectorized float4 path)
    const float4 gp = *reinterpret_cast<const float4*>(g_pair + lane * 4);
    const float4 bp = *reinterpret_cast<const float4*>(b_pair + lane * 4);

    // stage 1: gather pair rows, LayerNorm over 128, pack f16 A-tile.
    // one global_load_b128 per lane per row; one 8-byte LDS store per lane per row.
    for (int e = 0; e < 16; ++e) {
        int src = sMI[wid][e][0], dst = sMI[wid][e][1];
        const float* pr = pair + ((size_t)src * Lres + (size_t)dst) * DPAIR;
        if (e < 15) {   // prefetch next gathered row (global_prefetch_b8)
            int s2 = sMI[wid][e + 1][0], d2 = sMI[wid][e + 1][1];
            __builtin_prefetch(pair + ((size_t)s2 * Lres + (size_t)d2) * DPAIR, 0, 0);
        }
        const float4 pv = *reinterpret_cast<const float4*>(pr + lane * 4);
        float mu  = wredf(pv.x + pv.y + pv.z + pv.w) * (1.f / 128.f);
        float d0 = pv.x - mu, d1 = pv.y - mu, d2 = pv.z - mu, d3 = pv.w - mu;
        float var = wredf(d0 * d0 + d1 * d1 + d2 * d2 + d3 * d3) * (1.f / 128.f);
        float rs  = rsqrtf(var + EPSLN);
        v4h hv;
        hv[0] = (_Float16)(d0 * rs * gp.x + bp.x);
        hv[1] = (_Float16)(d1 * rs * gp.y + bp.y);
        hv[2] = (_Float16)(d2 * rs * gp.z + bp.z);
        hv[3] = (_Float16)(d3 * rs * gp.w + bp.w);
        *reinterpret_cast<v4h*>(&sA[wid][e * 128 + lane * 4]) = hv;
    }

    // WMMA stage 1: (16x128) x (128x32) -> y
    v8f c0 = {}, c1 = {};
#pragma unroll
    for (int kc = 0; kc < 4; ++kc) {
        v16h af  = load_a_frag(&sA[wid][0], lane, kc);
        v16h bf0 = load_bT_frag(sWe1T, lane, kc, 0);
        v16h bf1 = load_bT_frag(sWe1T, lane, kc, 16);
        c0 = __builtin_amdgcn_wmma_f32_16x16x32_f16(false, af, false, bf0, (short)0, c0, false, false);
        c1 = __builtin_amdgcn_wmma_f32_16x16x32_f16(false, af, false, bf1, (short)0, c1, false, false);
    }

    // row-LN over EF=32 entirely in registers: C VGPR r holds rows M=r (lanes 0-15)
    // and M=r+8 (lanes 16-31); half-wave allreduce handles both rows at once.
    // Normalized w_e written straight into inv A-tile slice K=64..95.
    {
        float be0 = be1[col], be16 = be1[col + 16];
        float ga  = g_e1[col], gb = g_e1[col + 16];
        float ba  = b_e1[col], bb = b_e1[col + 16];
        int rbase = hi << 3;
#pragma unroll
        for (int r = 0; r < 8; ++r) {
            int row = r + rbase;
            float y0 = c0[r] + be0;
            float y1 = c1[r] + be16;
            float mu = hredf(y0 + y1) * (1.f / 32.f);
            float d0 = y0 - mu, d1 = y1 - mu;
            float var = hredf(d0 * d0 + d1 * d1) * (1.f / 32.f);
            float rs = rsqrtf(var + EPSLN);
            sA[wid][row * 128 + 64 + col]      = (_Float16)(d0 * rs * ga + ba);
            sA[wid][row * 128 + 64 + col + 16] = (_Float16)(d1 * rs * gb + bb);
        }
    }

    // fill remaining inv slices: K=0..63 (l0 src/dst), K=96..127 (dn, sv, zero pad)
    for (int e = 0; e < 16; ++e) {
        int src = sMI[wid][e][0], dst = sMI[wid][e][1];
        _Float16* ar = &sA[wid][e * 128];
        ar[lane]      = (_Float16)l0[src * 32 + lane];
        ar[32 + lane] = (_Float16)l0[dst * 32 + lane];
        float tail = 0.f;
        if (lane == 0)      tail = sMF[wid][e][3];          // dn
        else if (lane < 4)  tail = sMF[wid][e][3 + lane];   // sv0..2
        ar[96 + lane] = (_Float16)tail;
    }

    // WMMA stage 2: inv @ W1, then b1 + gelu on accumulator regs
    v8f m0 = {}, m1 = {};
#pragma unroll
    for (int kc = 0; kc < 4; ++kc) {
        v16h af  = load_a_frag(&sA[wid][0], lane, kc);
        v16h bf0 = load_bT_frag(sW1T, lane, kc, 0);
        v16h bf1 = load_bT_frag(sW1T, lane, kc, 16);
        m0 = __builtin_amdgcn_wmma_f32_16x16x32_f16(false, af, false, bf0, (short)0, m0, false, false);
        m1 = __builtin_amdgcn_wmma_f32_16x16x32_f16(false, af, false, bf1, (short)0, m1, false, false);
    }
    {
        int rbase = hi << 3;
        float bb0 = b1w[col], bb1 = b1w[col + 16];
#pragma unroll
        for (int r = 0; r < 8; ++r) {
            sT[wid][(r + rbase) * 32 + col]      = gelu_tanh(m0[r] + bb0);
            sT[wid][(r + rbase) * 32 + col + 16] = gelu_tanh(m1[r] + bb1);
        }
    }

    // scatter: agg[dst] += m ; v_agg[dst] += cd*d_e + cv*(Wvv^T v[src])
    float wv0 = Wv[lane * 6 + 0], wv1 = Wv[lane * 6 + 1], wv2 = Wv[lane * 6 + 2];
    float wv3 = Wv[lane * 6 + 3], wv4 = Wv[lane * 6 + 4], wv5 = Wv[lane * 6 + 5];
    for (int e = 0; e < 16; ++e) {
        int src = sMI[wid][e][0], dst = sMI[wid][e][1];
        float mval = sT[wid][e * 32 + lane];
        atomicAdd(&agg[dst * 32 + lane], mval);
        float cf0 = wredf(mval * wv0);
        float cf1 = wredf(mval * wv1);
        float cf2 = wredf(mval * wv2);
        float cf3 = wredf(mval * wv3);
        float cf4 = wredf(mval * wv4);
        float cf5 = wredf(mval * wv5);
        if (lane < 9) {
            int o = lane / 3, x = lane % 3;
            float cd = (o == 0) ? cf0 : (o == 1) ? cf1 : cf2;
            float cv = (o == 0) ? cf3 : (o == 1) ? cf4 : cf5;
            const float* vs = vv + src * 9;
            float vp = Wvv[0 * 3 + o] * vs[0 * 3 + x]
                     + Wvv[1 * 3 + o] * vs[1 * 3 + x]
                     + Wvv[2 * 3 + o] * vs[2 * 3 + x];
            float vm = cd * sMF[wid][e][x] + cv * vp;
            atomicAdd(&v_agg[dst * 9 + lane], vm);
        }
    }
}

// ---------------- kernel 6: node outputs ----------------
__global__ __launch_bounds__(128)
void out_kernel(const float* __restrict__ l0, const float* __restrict__ agg,
                const float* __restrict__ vv, const float* __restrict__ v_agg,
                const float* __restrict__ CA,
                const float* __restrict__ W2, const float* __restrict__ b2,
                const float* __restrict__ Wself,
                const float* __restrict__ g_state, const float* __restrict__ b_state,
                const float* __restrict__ Wl, const float* __restrict__ bl,
                float* __restrict__ out_xyz, float* __restrict__ out_lddt) {
    __shared__ float sOff[4][9];
    int t = threadIdx.x, lane = t & 31, wid = t >> 5;
    int n = blockIdx.x * 4 + wid;

    // h_out = [l0, agg] @ W2 + b2  (shuffle-broadcast GEMV)
    float myl0 = l0[n * 32 + lane], myag = agg[n * 32 + lane];
    float h = b2[lane];
#pragma unroll
    for (int k = 0; k < 32; ++k) {
        h += __shfl(myl0, k, 32) * W2[k * 32 + lane];
        h += __shfl(myag, k, 32) * W2[(32 + k) * 32 + lane];
    }

    // v_out = Wself^T v + v_agg
    if (lane < 9) {
        int o = lane / 3, x = lane % 3;
        const float* vn = vv + n * 9;
        float vo = v_agg[n * 9 + lane]
                 + Wself[0 * 3 + o] * vn[0 * 3 + x]
                 + Wself[1 * 3 + o] * vn[1 * 3 + x]
                 + Wself[2 * 3 + o] * vn[2 * 3 + x];
        sOff[wid][lane] = vo;
    }
    __syncthreads();
    if (lane < 9) {
        int x = lane % 3;
        float ca_new = CA[n * 3 + x] + sOff[wid][3 + x];   // offset row o=1
        float base = ((n % Lres) == 0) ? 0.f : ca_new;     // residue 0 zeroed
        out_xyz[n * 9 + lane] = sOff[wid][lane] + base;
    }

    // lddt = sigmoid( LN(h) @ Wl + bl )
    float mu = wredf(h) * (1.f / 32.f);
    float d = h - mu;
    float var = wredf(d * d) * (1.f / 32.f);
    float lnh = d * rsqrtf(var + EPSLN) * g_state[lane] + b_state[lane];
    float s = wredf(lnh * Wl[lane]);
    if (lane == 0) out_lddt[n] = 1.f / (1.f + expf(-(s + bl[0])));
}

// ---------------- launch ----------------
extern "C" void kernel_launch(void* const* d_in, const int* in_sizes, int n_in,
                              void* d_out, int out_size, void* d_ws, size_t ws_size,
                              hipStream_t stream) {
    const float* xyz     = (const float*)d_in[0];
    const float* state   = (const float*)d_in[1];
    const float* msa     = (const float*)d_in[4];
    const float* pair    = (const float*)d_in[5];
    const float* seq1hot = (const float*)d_in[6];
    const float* g_msa   = (const float*)d_in[10];
    const float* b_msa   = (const float*)d_in[11];
    const float* g_pair  = (const float*)d_in[12];
    const float* b_pair  = (const float*)d_in[13];
    const float* g_state = (const float*)d_in[14];
    const float* b_state = (const float*)d_in[15];
    const float* g_node  = (const float*)d_in[16];
    const float* b_node  = (const float*)d_in[17];
    const float* g_e1    = (const float*)d_in[18];
    const float* b_e1    = (const float*)d_in[19];
    const float* Wx      = (const float*)d_in[20];
    const float* bx      = (const float*)d_in[21];
    const float* We1     = (const float*)d_in[22];
    const float* be1     = (const float*)d_in[23];
    const float* W1      = (const float*)d_in[24];
    const float* b1      = (const float*)d_in[25];
    const float* W2      = (const float*)d_in[26];
    const float* b2      = (const float*)d_in[27];
    const float* Wv      = (const float*)d_in[28];
    const float* Wvv     = (const float*)d_in[29];
    const float* Wself   = (const float*)d_in[30];
    const float* Wl      = (const float*)d_in[31];
    const float* bl      = (const float*)d_in[32];

    // scratch layout (floats)
    float* ws    = (float*)d_ws;
    float* l0    = ws;                         // 512*32  = 16384
    float* CA    = l0 + NNODE * 32;            // 512*3   = 1536
    float* vv    = CA + NNODE * 3;             // 512*9   = 4608
    float* agg   = vv + NNODE * 9;             // 512*32  = 16384
    float* v_agg = agg + NNODE * 32;           // 512*9   = 4608
    int*   nbr   = (int*)(v_agg + NNODE * 9);  // 512*64  ints

    float* out_xyz  = (float*)d_out;           // 4608
    float* out_lddt = out_xyz + NNODE * 9;     // 512

    prep_kernel<<<(NNODE + 255) / 256, 256, 0, stream>>>(xyz, CA, vv);
    node_kernel<<<NNODE, 256, 0, stream>>>(msa, seq1hot, state,
                                           g_msa, b_msa, g_state, b_state,
                                           g_node, b_node, Wx, bx, l0);
    topk_kernel<<<Lres, 256, 0, stream>>>(CA, nbr);
    zero_kernel<<<(NNODE * (32 + 9) + 255) / 256, 256, 0, stream>>>(agg, NNODE * (32 + 9));
    edge_kernel<<<NEDGE / (WPB * 16), 128, 0, stream>>>(pair, g_pair, b_pair,
                                                        We1, be1, g_e1, b_e1,
                                                        W1, b1, Wv, Wvv,
                                                        l0, CA, vv, nbr, agg, v_agg);
    out_kernel<<<NNODE / 4, 128, 0, stream>>>(l0, agg, vv, v_agg, CA,
                                              W2, b2, Wself, g_state, b_state,
                                              Wl, bl, out_xyz, out_lddt);
}
